// KAN_Convolutional_Layer_22256520528663
// MI455X (gfx1250) — matrix operand order repaired and checked by reading, not compiled, not run on hardware
//
#include <hip/hip_runtime.h>
#include <hip/hip_bf16.h>
#include <stdint.h>

// Problem constants (match reference)
#define CIN   32
#define COUT  64
#define HH    64
#define WW    64
#define PP    9          // K*K patch size
#define HO    62
#define WO    62
#define LLSZ  (HO * WO)          // 3844
#define NTOT  (16 * LLSZ)        // 61504 positions (multiple of 64)
#define KCH   96                 // per-channel K padded: 9*(1+8)=81 -> 96 (3 chunks of 32)
#define KTOT  (CIN * KCH)        // 3072
#define LDSROW 104               // bf16 elems per LDS row (208B: 16B-aligned, conflict-free)

typedef __attribute__((ext_vector_type(16))) __bf16 v16bf;
typedef __attribute__((ext_vector_type(8)))  float  v8f;

struct Frag32 { uint4 lo, hi; };   // 32 bytes == v16bf

static __device__ __forceinline__ unsigned short f2bf(float f) {
  unsigned int u = __builtin_bit_cast(unsigned int, f);
  u += 0x7FFFu + ((u >> 16) & 1u);          // round-to-nearest-even
  return (unsigned short)(u >> 16);
}

// ---------------------------------------------------------------------------
// Kernel 1: fold spline_scaler into spline_weight and pack combined bf16
// weight matrix Wc[COUT][KTOT], K-order: k = c*96 + p*9 + j
//   j==0   -> base_weight[o][c][p]          (pairs with silu feature)
//   j=1..8 -> spline_weight[o][c][p][j-1] * spline_scaler[o][c][p]
//   k in [81,96) per channel -> 0 padding
// ---------------------------------------------------------------------------
__global__ void kan_prep_weights(const float* __restrict__ bw,
                                 const float* __restrict__ sw,
                                 const float* __restrict__ ss,
                                 unsigned short* __restrict__ wc) {
  int tid = blockIdx.x * blockDim.x + threadIdx.x;
  if (tid >= COUT * CIN) return;
  int o = tid / CIN, c = tid % CIN;
  unsigned short* row = wc + (size_t)o * KTOT + c * KCH;
  #pragma unroll
  for (int p = 0; p < PP; ++p) {
    int base = (o * CIN + c) * PP + p;
    float scale = ss[base];
    row[p * 9 + 0] = f2bf(bw[base]);
    #pragma unroll
    for (int g = 0; g < 8; ++g)
      row[p * 9 + 1 + g] = f2bf(sw[base * 8 + g] * scale);
  }
  #pragma unroll
  for (int k = 81; k < KCH; ++k) row[k] = 0;
}

// ---------------------------------------------------------------------------
// Kernel 2: fused feature-gen + GEMM.
// Block = 256 threads (8 wave32), owns 64 positions x all 64 COUT.
// Wave w: N-subtile (w&3), M-tiles {2*(w>>2), 2*(w>>2)+1}.
// Feature stage: 192 threads, each owns one kernel-row (3 contiguous patch
// values) of one position; uniform-grid cubic B-spline evaluated in closed
// form (4 nonzero bases) and scattered into LDS.
// ---------------------------------------------------------------------------
__global__ __launch_bounds__(256)
void kan_fused(const float* __restrict__ x,
               const unsigned short* __restrict__ wc,
               float* __restrict__ out) {
  __shared__ alignas(16) unsigned short smA[64 * LDSROW + 8];
  unsigned short* dump = &smA[64 * LDSROW];   // sink for out-of-range scatter

  const int tid  = threadIdx.x;
  const int lane = tid & 31;
  const int wave = tid >> 5;
  const int nsub = wave & 3;          // which 16-position sub-tile
  const int m0   = (wave >> 2) * 2;   // first of two 16-COUT tiles
  const int nbase = blockIdx.x * 64;

  v8f acc0 = {};
  v8f acc1 = {};

  // zero the K-pad columns [81,96) once (constant across channels)
  for (int idx = tid; idx < 64 * 15; idx += 256) {
    int pos = idx / 15, col = 81 + idx % 15;
    smA[pos * LDSROW + col] = 0;
  }

  // ---- hoisted feature-thread setup (invariant over channel loop) ----
  const bool featThread = tid < 192;
  const float* xrow = x;              // safe dummy init
  unsigned short* fdst0 = dump;
  if (featThread) {
    int fpos  = tid & 63;             // position within block tile
    int ftrip = tid >> 6;             // kernel row py = 0..2
    int n  = nbase + fpos;
    int b  = n / LLSZ;  int rem = n - b * LLSZ;
    int oy = rem / WO;  int ox  = rem - oy * WO;
    xrow  = x + (((size_t)b * CIN) * HH + (oy + ftrip)) * WW + ox;
    fdst0 = smA + fpos * LDSROW + ftrip * 27;   // 3 p's of 9 feats each
  }

  // per-lane fragment addressing (16-bit WMMA operand layouts)
  const int r16    = lane & 15;
  const int hiHalf = lane >> 4;
  // A (16x32, rows=positions): lane<16 holds K {0..7,16..23}, lane>=16 {8..15,24..31}
  const unsigned short* aBase =
      smA + (nsub * 16 + r16) * LDSROW + hiHalf * 8;
  // B (32x16, cols=COUT): lane<16 holds K {0..7,16..23}, lane>=16 {8..15,24..31}
  const unsigned short* b0Base =
      wc + (size_t)(m0 * 16 + r16) * KTOT + hiHalf * 8;
  const unsigned short* b1Base = b0Base + (size_t)16 * KTOT;

  for (int c = 0; c < CIN; ++c) {
    __syncthreads();                  // previous chunk's A reads done
    // ---- stage features for this channel ----
    if (featThread) {
      const float* xp = xrow + (size_t)c * (HH * WW);
      #pragma unroll
      for (int k = 0; k < 3; ++k) {
        float v = xp[k];              // 3 contiguous loads -> b96
        unsigned short* dst = fdst0 + k * 9;
        // silu feature
        float sig = 1.0f / (1.0f + __expf(-v));
        dst[0] = f2bf(v * sig);
        // uniform-grid cubic B-spline: g0=-2.2, h=0.4, interval i=floor(t)
        float tt  = (v + 2.2f) * 2.5f;
        float fi  = floorf(tt);
        float t   = tt - fi;
        float omt = 1.0f - t;
        float w0 = omt * omt * omt * (1.0f / 6.0f);                    // N_{i-3}
        float w1 = ((3.0f * t - 6.0f) * t * t + 4.0f) * (1.0f / 6.0f); // N_{i-2}
        float w2 = (((-3.0f * t + 3.0f) * t + 3.0f) * t + 1.0f) *
                   (1.0f / 6.0f);                                       // N_{i-1}
        float w3 = t * t * t * (1.0f / 6.0f);                           // N_i
        int ii = (tt >= 0.0f && tt < 11.0f) ? (int)fi : -100;
        // zero 8 base slots, then scatter the 4 nonzero bases (j = ii-3..ii,
        // kept only if 0 <= j < 8; invalid lands in dump slot)
        #pragma unroll
        for (int z = 0; z < 8; ++z) dst[1 + z] = 0;
        float wv[4] = {w0, w1, w2, w3};
        #pragma unroll
        for (int s = 0; s < 4; ++s) {
          int j = ii - 3 + s;
          unsigned short* sp = ((unsigned)j < 8u) ? (dst + 1 + j) : dump;
          *sp = f2bf(wv[s]);
        }
      }
    }
    __syncthreads();                  // features visible to all waves

    // ---- GEMM: 3 chunks of K=32 for this channel ----
    const unsigned short* bw0 = b0Base + c * KCH;
    const unsigned short* bw1 = b1Base + c * KCH;
    #pragma unroll
    for (int kc = 0; kc < 3; ++kc) {
      const int k0 = kc * 32;
      Frag32 fa, fb0, fb1;
      fa.lo  = *(const uint4*)(aBase + k0);        // K half 0
      fa.hi  = *(const uint4*)(aBase + k0 + 16);   // K half 1
      fb0.lo = *(const uint4*)(bw0 + k0);
      fb0.hi = *(const uint4*)(bw0 + k0 + 16);
      fb1.lo = *(const uint4*)(bw1 + k0);
      fb1.hi = *(const uint4*)(bw1 + k0 + 16);
      v16bf a  = __builtin_bit_cast(v16bf, fa);
      v16bf b0 = __builtin_bit_cast(v16bf, fb0);
      v16bf b1 = __builtin_bit_cast(v16bf, fb1);
      acc0 = __builtin_amdgcn_wmma_f32_16x16x32_bf16(
                 false, a, false, b0, (short)0, acc0, false, false);
      acc1 = __builtin_amdgcn_wmma_f32_16x16x32_bf16(
                 false, a, false, b1, (short)0, acc1, false, false);
    }
  }

  // ---- epilogue: C/D layout -> out[b][o][oy][ox] (f32) ----
  // element (r,lane): M(row/position) = r + 8*(lane>=16), N(cout) = lane&15
  const int posBase = nsub * 16 + (hiHalf << 3);
  const int ocol = r16;
  #pragma unroll
  for (int r = 0; r < 8; ++r) {
    int n  = nbase + posBase + r;
    int b  = n / LLSZ;  int rem = n - b * LLSZ;
    int oy = rem / WO;  int ox  = rem - oy * WO;
    float* orow = out + (((size_t)b * COUT) * HO + oy) * WO + ox;
    orow[(size_t)(m0 * 16 + ocol)       * (HO * WO)] = acc0[r];
    orow[(size_t)((m0 + 1) * 16 + ocol) * (HO * WO)] = acc1[r];
  }
}

// ---------------------------------------------------------------------------
extern "C" void kernel_launch(void* const* d_in, const int* in_sizes, int n_in,
                              void* d_out, int out_size, void* d_ws, size_t ws_size,
                              hipStream_t stream) {
  const float* x  = (const float*)d_in[0];   // (16,32,64,64)
  const float* bw = (const float*)d_in[1];   // (64,32,9)
  const float* sw = (const float*)d_in[2];   // (64,32,9,8)
  const float* ss = (const float*)d_in[3];   // (64,32,9)
  float* out = (float*)d_out;                // (16,64,62,62)
  unsigned short* wc = (unsigned short*)d_ws; // 64*3072 bf16 = 384 KiB

  kan_prep_weights<<<(COUT * CIN + 255) / 256, 256, 0, stream>>>(bw, sw, ss, wc);
  kan_fused<<<NTOT / 64, 256, 0, stream>>>(x, wc, out);
}
